// DGCNNEncoderGn_46076409152164
// MI455X (gfx1250) — matrix operand order repaired and checked by reading, not compiled, not run on hardware
//
#include <hip/hip_runtime.h>
#include <hip/hip_bf16.h>

typedef __attribute__((ext_vector_type(16))) __bf16        v16bf;
typedef __attribute__((ext_vector_type(8)))  float         v8f;
typedef __attribute__((ext_vector_type(4)))  unsigned int  v4u;
typedef __attribute__((ext_vector_type(8)))  int           v8i;
typedef __attribute__((ext_vector_type(4)))  int           v4i;

#define BATCH 4
#define NPTS  1024
#define KNN   80
#define EPSQ  1e-5f
#define NEG_BIG (-3.402823466e38f)

// ---------- helpers ----------
__device__ __forceinline__ unsigned short f2bf(float f) {
  unsigned int u = __float_as_uint(f);
  unsigned int r = (u + 0x7FFFu + ((u >> 16) & 1u)) >> 16;   // RNE
  return (unsigned short)r;
}

union frag_u { v16bf v; uint4 q[2]; };

// Feature matrix is stored k-block swizzled: elem (k, n) at
//   Fb[ ((k>>5)*NPTS + n)*32 + (k&31) ]
// so per-lane fragment data is contiguous (two b128 loads per fragment).

// A fragment (16x32 bf16) from swizzled features: A[m][k] = F[k0+k, n0+m]
__device__ __forceinline__ v16bf load_frag_a_sw(const unsigned short* base /* + (kb*NPTS+n0)*32 */) {
  int lane = threadIdx.x & 31;
  int m = lane & 15;
  int khi = (lane >= 16) ? 8 : 0;
  const unsigned short* row = base + (size_t)m * 32;
  frag_u u;
  u.q[0] = *(const uint4*)(row + khi);
  u.q[1] = *(const uint4*)(row + 16 + khi);
  return u.v;
}
// A fragment from row-major weights: A[m][k] = W[(o0+m)*ldk + k0 + k]
__device__ __forceinline__ v16bf load_frag_a_w(const unsigned short* w /* + o0*ldk + k0 */, int ldk) {
  int lane = threadIdx.x & 31;
  int m = lane & 15;
  int khi = (lane >= 16) ? 8 : 0;
  const unsigned short* row = w + (size_t)m * ldk;
  frag_u u;
  u.q[0] = *(const uint4*)(row + khi);
  u.q[1] = *(const uint4*)(row + 16 + khi);
  return u.v;
}
// B fragment (32x16 bf16) from swizzled features: B[k][j] = F[k0+k, n0+j]
__device__ __forceinline__ v16bf load_frag_b_sw(const unsigned short* base /* + (kb*NPTS+n0)*32 */) {
  int lane = threadIdx.x & 31;
  int j = lane & 15;
  int kh = (lane >= 16) ? 8 : 0;
  const unsigned short* row = base + (size_t)j * 32 + kh;
  frag_u u;
  u.q[0] = *(const uint4*)(row);
  u.q[1] = *(const uint4*)(row + 16);
  return u.v;
}

__device__ __forceinline__ unsigned lds_offset_of(void* p) {
  return (unsigned)(unsigned long long)(__attribute__((address_space(3))) void*)p;
}

__device__ __forceinline__ v8f wmma_bf16(v16bf a, v16bf b, v8f c) {
  return __builtin_amdgcn_wmma_f32_16x16x32_bf16(false, a, false, b, (short)0, c, false, false);
}

// ---------- conversion kernels ----------
// F: (B, Cin, N) fp32 (batch stride) -> swizzled bf16 (B, Cpad/32, N, 32), zero padded
__global__ void conv_feat_bf16(const float* __restrict__ F, unsigned short* __restrict__ Fb,
                               int batchStride, int Cin, int Cpad) {
  int gid = blockIdx.x * blockDim.x + threadIdx.x;
  int total = BATCH * Cpad * NPTS;
  if (gid >= total) return;
  int n = gid % NPTS;
  int c = (gid / NPTS) % Cpad;
  int b = gid / (NPTS * Cpad);
  float v = (c < Cin) ? F[(size_t)b * batchStride + c * NPTS + n] : 0.0f;
  Fb[(size_t)b * Cpad * NPTS + ((size_t)(c >> 5) * NPTS + n) * 32 + (c & 31)] = f2bf(v);
}

__global__ void xx_kernel(const float* __restrict__ F, float* __restrict__ xx,
                          int batchStride, int Cin) {
  int gid = blockIdx.x * blockDim.x + threadIdx.x;
  if (gid >= BATCH * NPTS) return;
  int n = gid % NPTS, b = gid / NPTS;
  float s = 0.0f;
  for (int c = 0; c < Cin; ++c) {
    float v = F[(size_t)b * batchStride + c * NPTS + n];
    s += v * v;
  }
  xx[gid] = s;
}

// conv_w (Cout, 2*Cin) -> W1b = bf16(w[:, :Cin]), W12b = bf16(w[:, Cin:] - w[:, :Cin]); row-major, padded
__global__ void conv_w_kernel(const float* __restrict__ w, unsigned short* __restrict__ W1b,
                              unsigned short* __restrict__ W12b, int Cout, int Cin, int Cpad) {
  int gid = blockIdx.x * blockDim.x + threadIdx.x;
  if (gid >= Cout * Cpad) return;
  int k = gid % Cpad, o = gid / Cpad;
  float w1 = 0.0f, w2 = 0.0f;
  if (k < Cin) { w1 = w[o * 2 * Cin + k]; w2 = w[o * 2 * Cin + Cin + k]; }
  W1b[o * Cpad + k]  = f2bf(w1);
  W12b[o * Cpad + k] = f2bf(w2 - w1);
}

__global__ void conv_plain_bf16(const float* __restrict__ src, unsigned short* __restrict__ dst, int count) {
  int gid = blockIdx.x * blockDim.x + threadIdx.x;
  if (gid < count) dst[gid] = f2bf(src[gid]);
}

// ---------- WMMA GEMM kernels ----------
// One wave computes a 16x64 strip (4 accumulators); k-loop fully unrolled
// (compile-time CPAD) and double-buffered so next step's fragment loads
// overlap the current step's WMMAs.

// Gram + kNN score: S[b][n][m] = 2 * sum_c F[c,n]F[c,m] - xx[m]
template <int CPAD>
__global__ void gram_score_kernel(const unsigned short* __restrict__ Fb, const float* __restrict__ xx,
                                  float* __restrict__ S) {
  constexpr int STEPS = CPAD / 32;
  int tile = blockIdx.x;                 // (N/16) row tiles * (N/64) col groups
  int b = blockIdx.y;
  int tn = tile / (NPTS / 64);
  int tg = tile % (NPTS / 64);
  const unsigned short* F = Fb + (size_t)b * CPAD * NPTS;
  v8f acc[4] = {};
  v16bf a_cur, b_cur[4];
  a_cur = load_frag_a_sw(F + (size_t)tn * 16 * 32);
#pragma unroll
  for (int i = 0; i < 4; ++i) b_cur[i] = load_frag_b_sw(F + (size_t)(tg * 64 + i * 16) * 32);
#pragma unroll
  for (int s = 0; s < STEPS; ++s) {
    v16bf a_nxt, b_nxt[4];
    if (s + 1 < STEPS) {
      const unsigned short* kb = F + (size_t)(s + 1) * NPTS * 32;
      a_nxt = load_frag_a_sw(kb + (size_t)tn * 16 * 32);
#pragma unroll
      for (int i = 0; i < 4; ++i) b_nxt[i] = load_frag_b_sw(kb + (size_t)(tg * 64 + i * 16) * 32);
    }
#pragma unroll
    for (int i = 0; i < 4; ++i) acc[i] = wmma_bf16(a_cur, b_cur[i], acc[i]);
    if (s + 1 < STEPS) {
      a_cur = a_nxt;
#pragma unroll
      for (int i = 0; i < 4; ++i) b_cur[i] = b_nxt[i];
    }
  }
  int lane = threadIdx.x & 31;
  float* Sb = S + (size_t)b * NPTS * NPTS;
#pragma unroll
  for (int i = 0; i < 4; ++i) {
    int mcol = tg * 64 + i * 16 + (lane & 15);
    float xm = xx[b * NPTS + mcol];
#pragma unroll
    for (int r = 0; r < 8; ++r) {
      int nrow = tn * 16 + r + ((lane >= 16) ? 8 : 0);
      Sb[(size_t)nrow * NPTS + mcol] = 2.0f * acc[i][r] - xm;
    }
  }
}

// OutT[(b*N+n)*Cout + o] = sum_k Wb[o][k] * F[b][k][n]
template <int CPAD>
__global__ void lin_gemm_kernel(const unsigned short* __restrict__ Wb, const unsigned short* __restrict__ Fb,
                                float* __restrict__ OutT, int Cout) {
  constexpr int STEPS = CPAD / 32;
  int tilesM = Cout / 16;
  int tile = blockIdx.x;                 // tilesM * (N/64)
  int b = blockIdx.y;
  int to = tile % tilesM;
  int tg = tile / tilesM;
  const unsigned short* F = Fb + (size_t)b * CPAD * NPTS;
  const unsigned short* W = Wb + (size_t)to * 16 * CPAD;
  v8f acc[4] = {};
  v16bf a_cur, b_cur[4];
  a_cur = load_frag_a_w(W, CPAD);
#pragma unroll
  for (int i = 0; i < 4; ++i) b_cur[i] = load_frag_b_sw(F + (size_t)(tg * 64 + i * 16) * 32);
#pragma unroll
  for (int s = 0; s < STEPS; ++s) {
    v16bf a_nxt, b_nxt[4];
    if (s + 1 < STEPS) {
      const unsigned short* kb = F + (size_t)(s + 1) * NPTS * 32;
      a_nxt = load_frag_a_w(W + (s + 1) * 32, CPAD);
#pragma unroll
      for (int i = 0; i < 4; ++i) b_nxt[i] = load_frag_b_sw(kb + (size_t)(tg * 64 + i * 16) * 32);
    }
#pragma unroll
    for (int i = 0; i < 4; ++i) acc[i] = wmma_bf16(a_cur, b_cur[i], acc[i]);
    if (s + 1 < STEPS) {
      a_cur = a_nxt;
#pragma unroll
      for (int i = 0; i < 4; ++i) b_cur[i] = b_nxt[i];
    }
  }
  int lane = threadIdx.x & 31;
#pragma unroll
  for (int i = 0; i < 4; ++i) {
    int n = tg * 64 + i * 16 + (lane & 15);
    float* dst = OutT + ((size_t)b * NPTS + n) * Cout;
#pragma unroll
    for (int r = 0; r < 8; ++r) {
      int o = to * 16 + r + ((lane >= 16) ? 8 : 0);
      dst[o] = acc[i][r];
    }
  }
}

// Y[b][co][n] = sum_k Wmlp[co][k]*xfeat_b[b][k][n] + bias[co]   (M=1024, K=256)
__global__ void mlp_gemm_kernel(const unsigned short* __restrict__ Wb, const unsigned short* __restrict__ Fb,
                                const float* __restrict__ bias, float* __restrict__ Y) {
  constexpr int STEPS = 8;               // K = 256
  int tile = blockIdx.x;                 // 64 row tiles * 16 col groups
  int b = blockIdx.y;
  int to = tile / 16;
  int tg = tile % 16;
  const unsigned short* F = Fb + (size_t)b * 256 * NPTS;
  const unsigned short* W = Wb + (size_t)to * 16 * 256;
  v8f acc[4] = {};
  v16bf a_cur, b_cur[4];
  a_cur = load_frag_a_w(W, 256);
#pragma unroll
  for (int i = 0; i < 4; ++i) b_cur[i] = load_frag_b_sw(F + (size_t)(tg * 64 + i * 16) * 32);
#pragma unroll
  for (int s = 0; s < STEPS; ++s) {
    v16bf a_nxt, b_nxt[4];
    if (s + 1 < STEPS) {
      const unsigned short* kb = F + (size_t)(s + 1) * NPTS * 32;
      a_nxt = load_frag_a_w(W + (s + 1) * 32, 256);
#pragma unroll
      for (int i = 0; i < 4; ++i) b_nxt[i] = load_frag_b_sw(kb + (size_t)(tg * 64 + i * 16) * 32);
    }
#pragma unroll
    for (int i = 0; i < 4; ++i) acc[i] = wmma_bf16(a_cur, b_cur[i], acc[i]);
    if (s + 1 < STEPS) {
      a_cur = a_nxt;
#pragma unroll
      for (int i = 0; i < 4; ++i) b_cur[i] = b_nxt[i];
    }
  }
  int lane = threadIdx.x & 31;
#pragma unroll
  for (int i = 0; i < 4; ++i) {
    int n = tg * 64 + i * 16 + (lane & 15);
#pragma unroll
    for (int r = 0; r < 8; ++r) {
      int o = to * 16 + r + ((lane >= 16) ? 8 : 0);
      Y[((size_t)b * 1024 + o) * NPTS + n] = acc[i][r] + bias[o];
    }
  }
}

// ---------- top-K selection; row staged into LDS by the Tensor Data Mover ----------
__global__ void topk_kernel(const float* __restrict__ S, int* __restrict__ idx) {
  __shared__ float sc[NPTS];
  __shared__ float rv[128];
  __shared__ int   ri[128];
  int n = blockIdx.x, b = blockIdx.y;
  int t = threadIdx.x;                       // 128 threads
  const float* row = S + ((size_t)b * NPTS + n) * NPTS;

  if (t < 32) {                              // wave 0 issues the TDM copy (4 KB, 1-D tile)
    unsigned long long ga = (unsigned long long)row;
    unsigned lds = lds_offset_of((void*)sc);
    v4u g0;
    g0[0] = 1u;                                               // count=1 valid descriptor
    g0[1] = lds;                                              // lds_addr (bytes)
    g0[2] = (unsigned)(ga & 0xFFFFFFFFull);                   // global_addr[31:0]
    g0[3] = (unsigned)((ga >> 32) & 0x1FFFFFFull) | (2u << 30); // global_addr[56:32] | type=2
    v8i g1;
    g1[0] = 0x20000;                 // workgroup_mask=0, data_size=2 (4 bytes)
    g1[1] = (int)(1024u << 16);      // tensor_dim0[15:0]=1024 in bits[31:16]
    g1[2] = 0;                       // tensor_dim0 hi, tensor_dim1 lo
    g1[3] = (int)(1024u << 16);      // tile_dim0=1024 in bits[31:16]
    g1[4] = 0;                       // tile_dim1/tile_dim2 unused
    g1[5] = 1024;                    // tensor_dim0_stride[31:0]
    g1[6] = 0;
    g1[7] = 0;
    v4i gz = {0, 0, 0, 0};
#if defined(__clang_major__) && (__clang_major__ >= 23)
    v8i gz8 = {0, 0, 0, 0, 0, 0, 0, 0};
    __builtin_amdgcn_tensor_load_to_lds(g0, g1, gz, gz, gz8, 0);
#else
    __builtin_amdgcn_tensor_load_to_lds(g0, g1, gz, gz, 0);
#endif
    __builtin_amdgcn_s_wait_tensorcnt(0);
  }
  __syncthreads();

  int* out = idx + ((size_t)b * NPTS + n) * KNN;
  for (int it = 0; it < KNN; ++it) {
    float best = NEG_BIG; int bi = 0;
    for (int m = t; m < NPTS; m += 128) {
      float v = sc[m];
      if (v > best) { best = v; bi = m; }
    }
    rv[t] = best; ri[t] = bi;
    __syncthreads();
    for (int s = 64; s > 0; s >>= 1) {
      if (t < s && rv[t + s] > rv[t]) { rv[t] = rv[t + s]; ri[t] = ri[t + s]; }
      __syncthreads();
    }
    if (t == 0) { out[it] = ri[0]; sc[ri[0]] = NEG_BIG; }
    __syncthreads();
  }
}

// ---------- edge aggregation: gather, per-(n,o) max/min, per-(n,group) sums ----------
__global__ void edge_agg_kernel(const float* __restrict__ A_T, const float* __restrict__ Bv_T,
                                const int* __restrict__ idx, float* __restrict__ Ymax,
                                float* __restrict__ Ymin, float* __restrict__ psum,
                                float* __restrict__ psq, int Cout) {
  __shared__ int   sid[KNN];
  __shared__ float red[128];
  int n = blockIdx.x, b = blockIdx.y;
  int o = threadIdx.x;                       // blockDim = Cout
  for (int i = o; i < KNN; i += blockDim.x) sid[i] = idx[((size_t)b * NPTS + n) * KNN + i];
  __syncthreads();
  float bv = Bv_T[((size_t)b * NPTS + n) * Cout + o];
  float mx = NEG_BIG, mn = -NEG_BIG, s = 0.0f, s2 = 0.0f;
  for (int kk = 0; kk < KNN; ++kk) {
    if (kk + 1 < KNN) __builtin_prefetch(&A_T[((size_t)b * NPTS + sid[kk + 1]) * Cout + o], 0, 0);
    float y = A_T[((size_t)b * NPTS + sid[kk]) * Cout + o] + bv;
    mx = fmaxf(mx, y); mn = fminf(mn, y);
    s += y; s2 += y * y;
  }
  Ymax[((size_t)b * NPTS + n) * Cout + o] = mx;
  Ymin[((size_t)b * NPTS + n) * Cout + o] = mn;
  int half = Cout >> 1;
  int lo = o & (half - 1);
  red[o] = s; __syncthreads();
  for (int st = half >> 1; st > 0; st >>= 1) {
    if (lo < st) red[o] += red[o + st];
    __syncthreads();
  }
  if (lo == 0) psum[((size_t)b * NPTS + n) * 2 + (o / half)] = red[o];
  __syncthreads();
  red[o] = s2; __syncthreads();
  for (int st = half >> 1; st > 0; st >>= 1) {
    if (lo < st) red[o] += red[o + st];
    __syncthreads();
  }
  if (lo == 0) psq[((size_t)b * NPTS + n) * 2 + (o / half)] = red[o];
}

// deterministic group-stat reduction over n
__global__ void gstat_reduce_kernel(const float* __restrict__ psum, const float* __restrict__ psq,
                                    float* __restrict__ gstats) {
  __shared__ float r1[256], r2[256];
  int bg = blockIdx.x;                  // b*2+g
  int b = bg >> 1, g = bg & 1;
  int t = threadIdx.x;
  float s = 0.0f, q = 0.0f;
  for (int n = t; n < NPTS; n += 256) {
    s += psum[((size_t)b * NPTS + n) * 2 + g];
    q += psq[((size_t)b * NPTS + n) * 2 + g];
  }
  r1[t] = s; r2[t] = q; __syncthreads();
  for (int st = 128; st > 0; st >>= 1) {
    if (t < st) { r1[t] += r1[t + st]; r2[t] += r2[t + st]; }
    __syncthreads();
  }
  if (t == 0) { gstats[bg * 2 + 0] = r1[0]; gstats[bg * 2 + 1] = r2[0]; }
}

// groupnorm affine + LeakyReLU + max-over-k via (ymax, ymin) extremes
__global__ void edge_final_kernel(const float* __restrict__ Ymax, const float* __restrict__ Ymin,
                                  const float* __restrict__ gstats, const float* __restrict__ gnw,
                                  const float* __restrict__ gnb, float* __restrict__ xfeat,
                                  int Cout, int chOff) {
  int gid = blockIdx.x * blockDim.x + threadIdx.x;
  int total = BATCH * Cout * NPTS;
  if (gid >= total) return;
  int n = gid % NPTS;
  int o = (gid / NPTS) % Cout;
  int b = gid / (NPTS * Cout);
  int g = o / (Cout / 2);
  float cnt = (float)(Cout / 2) * (float)NPTS * (float)KNN;
  float su = gstats[(b * 2 + g) * 2 + 0];
  float sq = gstats[(b * 2 + g) * 2 + 1];
  float mean = su / cnt;
  float var = sq / cnt - mean * mean;
  float r = rsqrtf(var + EPSQ);
  float scale = gnw[o] * r;
  float shift = gnb[o] - mean * scale;
  float a = scale * Ymax[((size_t)b * NPTS + n) * Cout + o] + shift;
  float c = scale * Ymin[((size_t)b * NPTS + n) * Cout + o] + shift;
  a = (a >= 0.0f) ? a : 0.2f * a;
  c = (c >= 0.0f) ? c : 0.2f * c;
  xfeat[(size_t)b * 256 * NPTS + (size_t)(chOff + o) * NPTS + n] = fmaxf(a, c);
}

// ---------- stage D reductions ----------
__global__ void colstats_kernel(const float* __restrict__ Y, float* __restrict__ comax,
                                float* __restrict__ comin, float* __restrict__ cosum,
                                float* __restrict__ cosum2) {
  __shared__ float r1[256], r2[256], r3[256], r4[256];
  int co = blockIdx.x, b = blockIdx.y;
  int t = threadIdx.x;
  const float* row = Y + ((size_t)b * 1024 + co) * NPTS;
  float mx = NEG_BIG, mn = -NEG_BIG, s = 0.0f, q = 0.0f;
  for (int n = t; n < NPTS; n += 256) {
    float v = row[n];
    mx = fmaxf(mx, v); mn = fminf(mn, v);
    s += v; q += v * v;
  }
  r1[t] = mx; r2[t] = mn; r3[t] = s; r4[t] = q;
  __syncthreads();
  for (int st = 128; st > 0; st >>= 1) {
    if (t < st) {
      r1[t] = fmaxf(r1[t], r1[t + st]);
      r2[t] = fminf(r2[t], r2[t + st]);
      r3[t] += r3[t + st];
      r4[t] += r4[t + st];
    }
    __syncthreads();
  }
  if (t == 0) {
    comax[b * 1024 + co] = r1[0];
    comin[b * 1024 + co] = r2[0];
    cosum[b * 1024 + co] = r3[0];
    cosum2[b * 1024 + co] = r4[0];
  }
}

__global__ void final_kernel(const float* __restrict__ comax, const float* __restrict__ comin,
                             const float* __restrict__ cosum, const float* __restrict__ cosum2,
                             const float* __restrict__ gnw, const float* __restrict__ gnb,
                             float* __restrict__ x4) {
  __shared__ float r1[128], r2[128];
  int bg = blockIdx.x;            // b*8+g
  int b = bg / 8, g = bg % 8;
  int t = threadIdx.x;            // 128
  int co = g * 128 + t;
  r1[t] = cosum[b * 1024 + co];
  r2[t] = cosum2[b * 1024 + co];
  __syncthreads();
  for (int st = 64; st > 0; st >>= 1) {
    if (t < st) { r1[t] += r1[t + st]; r2[t] += r2[t + st]; }
    __syncthreads();
  }
  float cnt = 128.0f * (float)NPTS;
  float mean = r1[0] / cnt;
  float var = r2[0] / cnt - mean * mean;
  float r = rsqrtf(var + EPSQ);
  float scale = gnw[co] * r;
  float shift = gnb[co] - mean * scale;
  float a = fmaxf(scale * comax[b * 1024 + co] + shift, 0.0f);
  float c = fmaxf(scale * comin[b * 1024 + co] + shift, 0.0f);
  x4[b * 1024 + co] = fmaxf(a, c);
}

// ---------- workspace layout (bytes) ----------
static constexpr size_t OFF_S   = 0;                          // score matrix / MLP Y (16 MB, reused)
static constexpr size_t OFF_FB  = OFF_S   + 16777216;         // bf16 swizzled features (B,256,N)
static constexpr size_t OFF_XX  = OFF_FB  + 2097152;
static constexpr size_t OFF_IDX = OFF_XX  + 16384;
static constexpr size_t OFF_W1  = OFF_IDX + 1310720;
static constexpr size_t OFF_W12 = OFF_W1  + 32768;
static constexpr size_t OFF_WM  = OFF_W12 + 32768;
static constexpr size_t OFF_AT  = OFF_WM  + 524288;
static constexpr size_t OFF_BT  = OFF_AT  + 2097152;
static constexpr size_t OFF_YMX = OFF_BT  + 2097152;
static constexpr size_t OFF_YMN = OFF_YMX + 2097152;
static constexpr size_t OFF_PS  = OFF_YMN + 2097152;
static constexpr size_t OFF_PQ  = OFF_PS  + 32768;
static constexpr size_t OFF_GS  = OFF_PQ  + 32768;
static constexpr size_t OFF_CMX = OFF_GS  + 256;
static constexpr size_t OFF_CMN = OFF_CMX + 16384;
static constexpr size_t OFF_CS  = OFF_CMN + 16384;
static constexpr size_t OFF_CQ  = OFF_CS  + 16384;

extern "C" void kernel_launch(void* const* d_in, const int* in_sizes, int n_in,
                              void* d_out, int out_size, void* d_ws, size_t ws_size,
                              hipStream_t stream) {
  (void)in_sizes; (void)n_in; (void)out_size; (void)ws_size;
  const float* x       = (const float*)d_in[0];
  const float* conv1_w = (const float*)d_in[1];
  const float* gn1_w   = (const float*)d_in[2];
  const float* gn1_b   = (const float*)d_in[3];
  const float* conv2_w = (const float*)d_in[4];
  const float* gn2_w   = (const float*)d_in[5];
  const float* gn2_b   = (const float*)d_in[6];
  const float* conv3_w = (const float*)d_in[7];
  const float* gn3_w   = (const float*)d_in[8];
  const float* gn3_b   = (const float*)d_in[9];
  const float* mlp1_w  = (const float*)d_in[10];
  const float* mlp1_b  = (const float*)d_in[11];
  const float* gn5_w   = (const float*)d_in[12];
  const float* gn5_b   = (const float*)d_in[13];

  float* out   = (float*)d_out;
  float* x4    = out;                 // (B,1024)
  float* xfeat = out + BATCH * 1024;  // (B,256,N)

  char* ws = (char*)d_ws;
  float*          S     = (float*)(ws + OFF_S);
  unsigned short* Fb    = (unsigned short*)(ws + OFF_FB);
  float*          xxp   = (float*)(ws + OFF_XX);
  int*            idxp  = (int*)(ws + OFF_IDX);
  unsigned short* W1b   = (unsigned short*)(ws + OFF_W1);
  unsigned short* W12b  = (unsigned short*)(ws + OFF_W12);
  unsigned short* Wmlp  = (unsigned short*)(ws + OFF_WM);
  float*          A_T   = (float*)(ws + OFF_AT);
  float*          Bv_T  = (float*)(ws + OFF_BT);
  float*          Ymax  = (float*)(ws + OFF_YMX);
  float*          Ymin  = (float*)(ws + OFF_YMN);
  float*          psum  = (float*)(ws + OFF_PS);
  float*          psq   = (float*)(ws + OFF_PQ);
  float*          gst   = (float*)(ws + OFF_GS);
  float*          comax = (float*)(ws + OFF_CMX);
  float*          comin = (float*)(ws + OFF_CMN);
  float*          cosum = (float*)(ws + OFF_CS);
  float*          cosq  = (float*)(ws + OFF_CQ);

  // ---- layer 1: input x (B,3,N), Cpad=32, Cout=64 -> x1 at channels [0,64)
  {
    conv_feat_bf16<<<(BATCH * 32 * NPTS + 255) / 256, 256, 0, stream>>>(x, Fb, 3 * NPTS, 3, 32);
    xx_kernel<<<(BATCH * NPTS + 255) / 256, 256, 0, stream>>>(x, xxp, 3 * NPTS, 3);
    gram_score_kernel<32><<<dim3((NPTS / 16) * (NPTS / 64), BATCH), 32, 0, stream>>>(Fb, xxp, S);
    topk_kernel<<<dim3(NPTS, BATCH), 128, 0, stream>>>(S, idxp);
    conv_w_kernel<<<(64 * 32 + 255) / 256, 256, 0, stream>>>(conv1_w, W1b, W12b, 64, 3, 32);
    lin_gemm_kernel<32><<<dim3((64 / 16) * (NPTS / 64), BATCH), 32, 0, stream>>>(W1b, Fb, A_T, 64);
    lin_gemm_kernel<32><<<dim3((64 / 16) * (NPTS / 64), BATCH), 32, 0, stream>>>(W12b, Fb, Bv_T, 64);
    edge_agg_kernel<<<dim3(NPTS, BATCH), 64, 0, stream>>>(A_T, Bv_T, idxp, Ymax, Ymin, psum, psq, 64);
    gstat_reduce_kernel<<<BATCH * 2, 256, 0, stream>>>(psum, psq, gst);
    edge_final_kernel<<<(BATCH * 64 * NPTS + 255) / 256, 256, 0, stream>>>(Ymax, Ymin, gst, gn1_w, gn1_b,
                                                                           xfeat, 64, 0);
  }
  // ---- layer 2: input x1, Cpad=64, Cout=64 -> x2 at channels [64,128)
  {
    conv_feat_bf16<<<(BATCH * 64 * NPTS + 255) / 256, 256, 0, stream>>>(xfeat, Fb, 256 * NPTS, 64, 64);
    xx_kernel<<<(BATCH * NPTS + 255) / 256, 256, 0, stream>>>(xfeat, xxp, 256 * NPTS, 64);
    gram_score_kernel<64><<<dim3((NPTS / 16) * (NPTS / 64), BATCH), 32, 0, stream>>>(Fb, xxp, S);
    topk_kernel<<<dim3(NPTS, BATCH), 128, 0, stream>>>(S, idxp);
    conv_w_kernel<<<(64 * 64 + 255) / 256, 256, 0, stream>>>(conv2_w, W1b, W12b, 64, 64, 64);
    lin_gemm_kernel<64><<<dim3((64 / 16) * (NPTS / 64), BATCH), 32, 0, stream>>>(W1b, Fb, A_T, 64);
    lin_gemm_kernel<64><<<dim3((64 / 16) * (NPTS / 64), BATCH), 32, 0, stream>>>(W12b, Fb, Bv_T, 64);
    edge_agg_kernel<<<dim3(NPTS, BATCH), 64, 0, stream>>>(A_T, Bv_T, idxp, Ymax, Ymin, psum, psq, 64);
    gstat_reduce_kernel<<<BATCH * 2, 256, 0, stream>>>(psum, psq, gst);
    edge_final_kernel<<<(BATCH * 64 * NPTS + 255) / 256, 256, 0, stream>>>(Ymax, Ymin, gst, gn2_w, gn2_b,
                                                                           xfeat, 64, 64);
  }
  // ---- layer 3: input x2, Cpad=64, Cout=128 -> x3 at channels [128,256)
  {
    conv_feat_bf16<<<(BATCH * 64 * NPTS + 255) / 256, 256, 0, stream>>>(xfeat + 64 * NPTS, Fb,
                                                                        256 * NPTS, 64, 64);
    xx_kernel<<<(BATCH * NPTS + 255) / 256, 256, 0, stream>>>(xfeat + 64 * NPTS, xxp, 256 * NPTS, 64);
    gram_score_kernel<64><<<dim3((NPTS / 16) * (NPTS / 64), BATCH), 32, 0, stream>>>(Fb, xxp, S);
    topk_kernel<<<dim3(NPTS, BATCH), 128, 0, stream>>>(S, idxp);
    conv_w_kernel<<<(128 * 64 + 255) / 256, 256, 0, stream>>>(conv3_w, W1b, W12b, 128, 64, 64);
    lin_gemm_kernel<64><<<dim3((128 / 16) * (NPTS / 64), BATCH), 32, 0, stream>>>(W1b, Fb, A_T, 128);
    lin_gemm_kernel<64><<<dim3((128 / 16) * (NPTS / 64), BATCH), 32, 0, stream>>>(W12b, Fb, Bv_T, 128);
    edge_agg_kernel<<<dim3(NPTS, BATCH), 128, 0, stream>>>(A_T, Bv_T, idxp, Ymax, Ymin, psum, psq, 128);
    gstat_reduce_kernel<<<BATCH * 2, 256, 0, stream>>>(psum, psq, gst);
    edge_final_kernel<<<(BATCH * 128 * NPTS + 255) / 256, 256, 0, stream>>>(Ymax, Ymin, gst, gn3_w, gn3_b,
                                                                            xfeat, 128, 128);
  }

  // ---- stage D: MLP 256 -> 1024, groupnorm(8), relu, max over N
  conv_feat_bf16<<<(BATCH * 256 * NPTS + 255) / 256, 256, 0, stream>>>(xfeat, Fb, 256 * NPTS, 256, 256);
  conv_plain_bf16<<<(1024 * 256 + 255) / 256, 256, 0, stream>>>(mlp1_w, Wmlp, 1024 * 256);
  mlp_gemm_kernel<<<dim3(64 * 16, BATCH), 32, 0, stream>>>(Wmlp, Fb, mlp1_b, S);
  colstats_kernel<<<dim3(1024, BATCH), 256, 0, stream>>>(S, comax, comin, cosum, cosq);
  final_kernel<<<BATCH * 8, 128, 0, stream>>>(comax, comin, cosum, cosq, gn5_w, gn5_b, x4);
}
